// ScaledDotProductAttention_30227979829451
// MI455X (gfx1250) — matrix operand order repaired and checked by reading, not compiled, not run on hardware
//
#include <hip/hip_runtime.h>
#include <hip/hip_fp16.h>

typedef _Float16 v16h __attribute__((ext_vector_type(16)));
typedef _Float16 v8h  __attribute__((ext_vector_type(8)));
typedef _Float16 v4h  __attribute__((ext_vector_type(4)));
typedef float    v8f  __attribute__((ext_vector_type(8)));
typedef float    v4f  __attribute__((ext_vector_type(4)));
typedef unsigned int u32x4 __attribute__((ext_vector_type(4)));
typedef int          i32x4 __attribute__((ext_vector_type(4)));
typedef int          i32x8 __attribute__((ext_vector_type(8)));

// D = A(16x32 f16) x B(32x16 f16) + C(16x16 f32)
static __device__ __forceinline__ v8f wmma_f16(v16h a, v16h b, v8f c) {
    return __builtin_amdgcn_wmma_f32_16x16x32_f16(false, a, false, b, (short)0, c, false, false);
}

// A-fragment (16-bit A 16x32, contiguous k order used by this code, matched by B): lane l holds row (l&15);
// elements K = {2kc..2kc+15} (contiguous), kc = (l>=16 ? 8 : 0). 16B chunks.
static __device__ __forceinline__ v16h make_afrag(const _Float16* rowPtr, int kc) {
    v8h lo = *(const v8h*)(rowPtr + 2 * kc);
    v8h hi = *(const v8h*)(rowPtr + 2 * kc + 8);
    v16h a;
#pragma unroll
    for (int i = 0; i < 8; ++i) { a[i] = lo[i]; a[8 + i] = hi[i]; }
    return a;
}

// A-fragment via 8B chunks (padded V rows in LDS are only 8B-aligned)
static __device__ __forceinline__ v16h make_afrag8(const _Float16* rowPtr, int kc) {
    v4h c0 = *(const v4h*)(rowPtr + 2 * kc);
    v4h c1 = *(const v4h*)(rowPtr + 2 * kc + 4);
    v4h c2 = *(const v4h*)(rowPtr + 2 * kc + 8);
    v4h c3 = *(const v4h*)(rowPtr + 2 * kc + 12);
    v16h a;
#pragma unroll
    for (int i = 0; i < 4; ++i) {
        a[i] = c0[i]; a[4 + i] = c1[i]; a[8 + i] = c2[i]; a[12 + i] = c3[i];
    }
    return a;
}

// B-fragment: 16 contiguous f16 (lane = column; j = K index khalf+j)
static __device__ __forceinline__ v16h make_bfrag(const _Float16* p) {
    v8h lo = *(const v8h*)p;
    v8h hi = *(const v8h*)(p + 8);
    v16h v;
#pragma unroll
    for (int i = 0; i < 8; ++i) { v[i] = lo[i]; v[8 + i] = hi[i]; }
    return v;
}

// ---------------------------------------------------------------------------
// Tensor Data Mover: 2D tile load Global -> LDS, D# per ISA §8.3/8.4.
// ---------------------------------------------------------------------------
static __device__ __forceinline__ void tdm_load_2d(
    unsigned lds_addr, const void* gaddr,
    unsigned tensor_d0, unsigned tensor_d1,
    unsigned tile_d0, unsigned tile_d1,
    unsigned long long stride0,
    unsigned pad_interval_code, unsigned pad_amount_code, bool pad_en)
{
    unsigned long long ga = (unsigned long long)(uintptr_t)gaddr;
    u32x4 g0;
    g0[0] = 1u;                                            // count = 1 (valid D#)
    g0[1] = lds_addr;                                      // LDS byte address
    g0[2] = (unsigned)(ga & 0xFFFFFFFFu);                  // global_addr[31:0]
    g0[3] = (unsigned)((ga >> 32) & 0x01FFFFFFu)           // global_addr[56:32]
          | 0x80000000u;                                   // type = 2 ("image")
    i32x8 g1;
    unsigned w0 = (1u << 16);                              // data_size = 2 bytes
    if (pad_en) w0 |= (1u << 20) | (pad_interval_code << 22) | (pad_amount_code << 25);
    g1[0] = (int)w0;
    g1[1] = (int)((tensor_d0 & 0xFFFFu) << 16);            // tensor_dim0[15:0]
    g1[2] = (int)(((tensor_d0 >> 16) & 0xFFFFu) | ((tensor_d1 & 0xFFFFu) << 16));
    g1[3] = (int)(((tensor_d1 >> 16) & 0xFFFFu) | ((tile_d0 & 0xFFFFu) << 16));
    g1[4] = (int)(tile_d1 & 0xFFFFu);                      // tile_dim1, tile_dim2=0
    g1[5] = (int)(stride0 & 0xFFFFFFFFull);                // tensor_dim0_stride[31:0]
    g1[6] = (int)((stride0 >> 32) & 0xFFFFull);            // stride0[47:32], stride1=0
    g1[7] = 0;
    i32x4 z4 = {0, 0, 0, 0};
#if __has_builtin(__builtin_amdgcn_tensor_load_to_lds)
#if __clang_major__ >= 23
    i32x8 z8 = {0, 0, 0, 0, 0, 0, 0, 0};
    __builtin_amdgcn_tensor_load_to_lds(g0, g1, z4, z4, z8, 0);
#else
    __builtin_amdgcn_tensor_load_to_lds(g0, g1, z4, z4, 0);
#endif
#else
    (void)g0; (void)g1; (void)z4;
#endif
}

// ---------------------------------------------------------------------------
__global__ void f32_to_f16_kernel(const float* __restrict__ src,
                                  _Float16* __restrict__ dst, int n) {
    int i = blockIdx.x * blockDim.x + threadIdx.x;
    if (i < n) dst[i] = (_Float16)src[i];
}

// ---------------------------------------------------------------------------
// Q/K projection: per batch Y(512,1024) = W(512,64) @ X(64,1024) + bias,
// stored per head as (n,64) rows; premul folds the 1/sqrt(8) logit scale into Q.
// grid (8 pgroups, 8 batch), block 256 (8 waves, one pixel-block each)
// ---------------------------------------------------------------------------
__global__ void __launch_bounds__(256)
qk_proj_kernel(const float* __restrict__ X,
               const _Float16* __restrict__ Wh,
               const float* __restrict__ bias,
               _Float16* __restrict__ dst, float premul)
{
    const int b    = blockIdx.y;
    const int wid  = threadIdx.x >> 5;
    const int lane = threadIdx.x & 31;
    const int pblock = blockIdx.x * 8 + wid;
    const int col  = lane & 15;
    const int hi16 = lane >> 4;
    const int p    = pblock * 16 + col;
    const int kc   = hi16 ? 8 : 0;
    const int rh   = hi16 * 8;
    const int khalf = hi16 * 16;

    const float* Xb = X + (size_t)b * 64 * 1024;

    v16h b0, b1;                     // B: X tile (64 x 16 px), lane-coalesced
#pragma unroll
    for (int j = 0; j < 16; ++j) {
        b0[j] = (_Float16)Xb[(khalf + j) * 1024 + p];
        b1[j] = (_Float16)Xb[(32 + khalf + j) * 1024 + p];
    }

#pragma unroll 4
    for (int mt = 0; mt < 32; ++mt) {
        const _Float16* Wrow = Wh + (mt * 16 + col) * 64;
        v16h a0 = make_afrag(Wrow, kc);
        v16h a1 = make_afrag(Wrow + 32, kc);
        v8f c = {};
        c = wmma_f16(a0, b0, c);
        c = wmma_f16(a1, b1, c);
        v8h vals;                    // rows = 8 consecutive per-head channels
#pragma unroll
        for (int r = 0; r < 8; ++r)
            vals[r] = (_Float16)((c[r] + bias[mt * 16 + r + rh]) * premul);
        const int och0 = mt * 16 + rh;
        const int head = och0 >> 6, cin0 = och0 & 63;
        *(v8h*)&dst[((size_t)(b * 8 + head) * 1024 + p) * 64 + cin0] = vals;
    }
}

// ---------------------------------------------------------------------------
// V projection, transposed orientation: Y^T = X^T W^T so the C tile is
// pixel-major per register -> packed 16B stores into the (b,h,64,n) layout.
// ---------------------------------------------------------------------------
__global__ void __launch_bounds__(256)
v_proj_kernel(const float* __restrict__ X,
              const _Float16* __restrict__ Wh,
              const float* __restrict__ bias,
              _Float16* __restrict__ dst)
{
    const int b    = blockIdx.y;
    const int wid  = threadIdx.x >> 5;
    const int lane = threadIdx.x & 31;
    const int pblock = blockIdx.x * 8 + wid;
    const int col  = lane & 15;
    const int hi16 = lane >> 4;
    const int kc   = hi16 ? 8 : 0;
    const int rh   = hi16 * 8;
    const int khalf = hi16 * 16;
    const int pbase = pblock * 16;

    const float* Xb   = X + (size_t)b * 64 * 1024;
    const float* Xcol = Xb + pbase + col;     // A row = pixel; element c strided

    v16h ax0, ax1;
#pragma unroll
    for (int j = 0; j < 8; ++j) {
        ax0[j]     = (_Float16)Xcol[(khalf + j) * 1024];
        ax0[8 + j] = (_Float16)Xcol[(khalf + 8 + j) * 1024];
        ax1[j]     = (_Float16)Xcol[(32 + khalf + j) * 1024];
        ax1[8 + j] = (_Float16)Xcol[(32 + khalf + 8 + j) * 1024];
    }

#pragma unroll 4
    for (int ot = 0; ot < 32; ++ot) {
        const _Float16* Wrow = Wh + (ot * 16 + col) * 64;   // B col = och
        v16h bw0 = *(const v16h*)(Wrow + khalf);
        v16h bw1 = *(const v16h*)(Wrow + 32 + khalf);
        v8f c = {};
        c = wmma_f16(ax0, bw0, c);
        c = wmma_f16(ax1, bw1, c);
        const int och = ot * 16 + col;
        const int head = och >> 6, cin = och & 63;
        const float bi = bias[och];
        v8h vals;                               // rows = 8 consecutive pixels
#pragma unroll
        for (int r = 0; r < 8; ++r) vals[r] = (_Float16)(c[r] + bi);
        *(v8h*)&dst[((size_t)(b * 8 + head) * 64 + cin) * 1024 + pbase + rh] = vals;
    }
}

// ---------------------------------------------------------------------------
// Flash attention, one workgroup per (b,h). TDM stages whole per-head
// K (1024x64, rows padded to 72 f16) and V (64x1024, rows padded to 1032 f16)
// into LDS; 8 waves each process 8 q-blocks out of LDS.
//
// Transposed logit tile S^T = K Q^T: rows = keys (VGPRs), cols = q (lanes).
// Softmax over keys is a per-lane register reduction + ONE shfl_xor(16)
// (lanes L and L+16 hold complementary key halves of the same q).
// O^T = V P^T: output tile is channel-major per register -> packed stores.
// grid (8 heads, 8 batch), block 256
// ---------------------------------------------------------------------------
__global__ void __launch_bounds__(256)
attn_kernel(const _Float16* __restrict__ qbuf,  // (b,h,n,64), Q pre-scaled
            const _Float16* __restrict__ kbuf,  // (b,h,n,64)
            const _Float16* __restrict__ vbuf,  // (b,h,64,n)
            _Float16* __restrict__ obuf)        // (b,n,512)
{
    __shared__ __align__(128) _Float16 Ksh[1024 * 72];
    __shared__ __align__(128) _Float16 Vsh[64 * 1032];
    __shared__ __align__(128) _Float16 Pt[8][16][40];   // per-wave P, q-major

    const int h = blockIdx.x, b = blockIdx.y;
    const int wid  = threadIdx.x >> 5;
    const int lane = threadIdx.x & 31;
    const int col  = lane & 15;
    const int hi16 = lane >> 4;
    const int kc   = hi16 ? 8 : 0;
    const int rh   = hi16 * 8;
    const int khalf = hi16 * 16;

    const _Float16* qhead = qbuf + (size_t)(b * 8 + h) * 1024 * 64;
    const _Float16* khead = kbuf + (size_t)(b * 8 + h) * 1024 * 64;
    const _Float16* vhead = vbuf + (size_t)(b * 8 + h) * 64 * 1024;

#if __has_builtin(__builtin_amdgcn_tensor_load_to_lds)
    if (wid == 0) {
        tdm_load_2d((unsigned)(uintptr_t)Ksh, khead, 64, 1024, 64, 1024, 64,
                    /*pad: 4 DW per 32 DW*/ 4u, 3u, true);
        tdm_load_2d((unsigned)(uintptr_t)Vsh, vhead, 1024, 64, 1024, 64, 1024,
                    /*pad: 2 DW per 256 DW*/ 7u, 1u, true);
        __builtin_amdgcn_s_wait_tensorcnt(0);
    }
#else
    for (int r0 = threadIdx.x; r0 < 1024; r0 += 256)
        for (int j = 0; j < 64; ++j)
            Ksh[r0 * 72 + j] = khead[(size_t)r0 * 64 + j];
    {
        int c = threadIdx.x >> 2, part = threadIdx.x & 3;
        int k0 = part * 256, bump = (k0 >= 512) ? 4 : 0;
        for (int j = 0; j < 256; ++j)
            Vsh[c * 1032 + k0 + bump + j] = vhead[(size_t)c * 1024 + k0 + j];
    }
#endif
    __syncthreads();

    for (int qi = 0; qi < 8; ++qi) {
        const int qblock = qi * 8 + wid;
        const int qbase  = qblock * 16;
        const int q      = qbase + col;

        // Q^T B-fragments (held across the key loop): lane = q column
        const _Float16* Qrow = qhead + (size_t)q * 64;
        v16h bq0 = *(const v16h*)(Qrow + khalf);
        v16h bq1 = *(const v16h*)(Qrow + 32 + khalf);

        v8f ovT[4] = {{}, {}, {}, {}};   // O^T: rows = channels, cols = q
        float m = -1e30f, l = 0.0f;

        for (int kb = 0; kb < 1024; kb += 32) {
            // K A-fragments: rows = keys, K-dim = channels
            const _Float16* K0 = &Ksh[(size_t)(kb + col) * 72];
            const _Float16* K1 = &Ksh[(size_t)(kb + 16 + col) * 72];
            v16h ak00 = make_afrag(K0, kc);
            v16h ak01 = make_afrag(K0 + 32, kc);
            v16h ak10 = make_afrag(K1, kc);
            v16h ak11 = make_afrag(K1 + 32, kc);

            v8f st0 = {}, st1 = {};
            st0 = wmma_f16(ak00, bq0, st0);
            st0 = wmma_f16(ak01, bq1, st0);
            st1 = wmma_f16(ak10, bq0, st1);
            st1 = wmma_f16(ak11, bq1, st1);
            // st0[r]: key kb + r + rh; st1[r]: key kb+16 + r + rh; col = q

            float sv[16];
#pragma unroll
            for (int r = 0; r < 8; ++r) { sv[r] = st0[r]; sv[8 + r] = st1[r]; }
            float mx = sv[0];
#pragma unroll
            for (int i = 1; i < 16; ++i) mx = fmaxf(mx, sv[i]);
            mx = fmaxf(mx, __shfl_xor(mx, 16));
            const float mnew  = fmaxf(m, mx);
            const float alpha = __expf(m - mnew);
            float rs = 0.0f;
#pragma unroll
            for (int i = 0; i < 16; ++i) {
                float pv = __expf(sv[i] - mnew);
                sv[i] = pv;
                rs += pv;
            }
            rs += __shfl_xor(rs, 16);
            l = l * alpha + rs;
            m = mnew;
#pragma unroll
            for (int ct = 0; ct < 4; ++ct)
#pragma unroll
                for (int r = 0; r < 8; ++r) ovT[ct][r] *= alpha;

            // stage P into q-major LDS rows (lane pair fills all 32 keys)
#pragma unroll
            for (int r = 0; r < 8; ++r) {
                Pt[wid][col][r + rh]      = (_Float16)sv[r];
                Pt[wid][col][16 + r + rh] = (_Float16)sv[8 + r];
            }

            // P^T B-fragment: lane = q column, contiguous keys (same-wave DS
            // ops are in-order, no barrier needed)
            v16h bp = make_bfrag(&Pt[wid][col][khalf]);

            // V A-fragments: rows = channels, K-dim = keys (padded rows)
            const int vkb = kb + ((kb >= 512) ? 4 : 0);
#pragma unroll
            for (int ct = 0; ct < 4; ++ct) {
                const _Float16* Vrow = &Vsh[(size_t)(ct * 16 + col) * 1032 + vkb];
                v16h av = make_afrag8(Vrow, kc);
                ovT[ct] = wmma_f16(av, bp, ovT[ct]);
            }
        }

        // normalize; rows of ovT are 8 consecutive channels -> packed stores
        const float inv = 1.0f / l;
#pragma unroll
        for (int ct = 0; ct < 4; ++ct) {
            v8h vals;
#pragma unroll
            for (int r = 0; r < 8; ++r) vals[r] = (_Float16)(ovT[ct][r] * inv);
            const int ch0 = h * 64 + ct * 16 + rh;
            *(v8h*)&obuf[((size_t)b * 1024 + q) * 512 + ch0] = vals;
        }
    }
}

// ---------------------------------------------------------------------------
// Output projection, transposed: out^T = O^T WO^T per batch.
// A = O^T rows (pixel-major, contiguous in (b,n,512)); B = WO^T (rows of WO).
// C tile rows = pixels -> two float4 stores per c-tile.
// grid (8 pgroups, 8 batch), block 256
// ---------------------------------------------------------------------------
__global__ void __launch_bounds__(256)
out_proj_kernel(const _Float16* __restrict__ obuf,
                const _Float16* __restrict__ WOh,
                const float* __restrict__ bO,
                float* __restrict__ out)
{
    const int b    = blockIdx.y;
    const int wid  = threadIdx.x >> 5;
    const int lane = threadIdx.x & 31;
    const int pblock = blockIdx.x * 8 + wid;
    const int col  = lane & 15;
    const int hi16 = lane >> 4;
    const int kc   = hi16 ? 8 : 0;
    const int rh   = hi16 * 8;
    const int khalf = hi16 * 16;
    const int pbase = pblock * 16;

    const _Float16* ob   = obuf + (size_t)b * 1024 * 512;
    const _Float16* obrw = ob + (size_t)(pbase + col) * 512;  // A row = pixel
    v8f acc[4] = {{}, {}, {}, {}};

    for (int k0 = 0; k0 < 512; k0 += 32) {
        v16h a = make_afrag(obrw + k0, kc);
#pragma unroll
        for (int ct = 0; ct < 4; ++ct) {
            v16h bw = *(const v16h*)(WOh + (size_t)(ct * 16 + col) * 512 + k0 + khalf);
            acc[ct] = wmma_f16(a, bw, acc[ct]);
        }
    }

#pragma unroll
    for (int ct = 0; ct < 4; ++ct) {
        const int cch = ct * 16 + col;
        const float bi = bO[cch];
        v4f lo, hi;
#pragma unroll
        for (int r = 0; r < 4; ++r) { lo[r] = acc[ct][r] + bi; hi[r] = acc[ct][4 + r] + bi; }
        float* dst = &out[((size_t)b * 64 + cch) * 1024 + pbase + rh];
        *(v4f*)dst       = lo;
        *(v4f*)(dst + 4) = hi;
    }
}

// ---------------------------------------------------------------------------
extern "C" void kernel_launch(void* const* d_in, const int* in_sizes, int n_in,
                              void* d_out, int out_size, void* d_ws, size_t ws_size,
                              hipStream_t stream) {
    const float* queries = (const float*)d_in[0];
    const float* keysx   = (const float*)d_in[1];
    const float* values  = (const float*)d_in[2];
    const float* WQ = (const float*)d_in[3];
    const float* bQ = (const float*)d_in[4];
    const float* WK = (const float*)d_in[5];
    const float* bK = (const float*)d_in[6];
    const float* WV = (const float*)d_in[7];
    const float* bV = (const float*)d_in[8];
    const float* WO = (const float*)d_in[9];
    const float* bO = (const float*)d_in[10];
    float* out = (float*)d_out;

    _Float16* ws   = (_Float16*)d_ws;
    _Float16* WQh  = ws;
    _Float16* WKh  = WQh + 512 * 64;
    _Float16* WVh  = WKh + 512 * 64;
    _Float16* WOh  = WVh + 512 * 64;
    _Float16* qbuf = WOh + 64 * 512;                       // (b,h,n,64)
    _Float16* kbuf = qbuf + (size_t)8 * 8 * 1024 * 64;     // (b,h,n,64)
    _Float16* vbuf = kbuf + (size_t)8 * 8 * 1024 * 64;     // (b,h,64,n)
    _Float16* obuf = vbuf + (size_t)8 * 8 * 64 * 1024;     // (b,n,512)

    const float scale = 0.35355339059327379f;              // 1/sqrt(8)

    f32_to_f16_kernel<<<128, 256, 0, stream>>>(WQ, WQh, 512 * 64);
    f32_to_f16_kernel<<<128, 256, 0, stream>>>(WK, WKh, 512 * 64);
    f32_to_f16_kernel<<<128, 256, 0, stream>>>(WV, WVh, 512 * 64);
    f32_to_f16_kernel<<<128, 256, 0, stream>>>(WO, WOh, 64 * 512);

    qk_proj_kernel<<<dim3(8, 8), 256, 0, stream>>>(queries, WQh, bQ, qbuf, scale);
    qk_proj_kernel<<<dim3(8, 8), 256, 0, stream>>>(keysx,   WKh, bK, kbuf, 1.0f);
    v_proj_kernel <<<dim3(8, 8), 256, 0, stream>>>(values,  WVh, bV, vbuf);

    attn_kernel<<<dim3(8, 8), 256, 0, stream>>>(qbuf, kbuf, vbuf, obuf);

    out_proj_kernel<<<dim3(8, 8), 256, 0, stream>>>(obuf, WOh, bO, out);
}
